// Net_31379031065089
// MI455X (gfx1250) — matrix operand order repaired and checked by reading, not compiled, not run on hardware
//
#include <hip/hip_runtime.h>
#include <hip/hip_bf16.h>

typedef __attribute__((ext_vector_type(16))) _Float16 v16h;
typedef __attribute__((ext_vector_type(8)))  float    v8f;

// ---------------------------------------------------------------------------
// Degree / normalization kernels
// ---------------------------------------------------------------------------
__global__ void k_init_deg(float* __restrict__ deg, int n) {
    int i = blockIdx.x * blockDim.x + threadIdx.x;
    if (i < n) deg[i] = 1.0f;                      // self-loop contributes 1
}

__global__ void k_edge_deg(const int* __restrict__ dst, float* __restrict__ deg, int e) {
    int i = blockIdx.x * blockDim.x + threadIdx.x;
    if (i < e) atomicAdd(&deg[dst[i]], 1.0f);
}

__global__ void k_rsqrt(float* __restrict__ deg, int n) {
    int i = blockIdx.x * blockDim.x + threadIdx.x;
    if (i < n) {
        float d = deg[i];
        deg[i] = (d > 0.0f) ? rsqrtf(d) : 0.0f;   // deg >= 1 always, but match ref
    }
}

// ---------------------------------------------------------------------------
// Pack weight matrix W [Kdim x Nout] (row stride ldw, f32) into WMMA B
// fragments (16-bit B 32x16 layout): fragment (colTile ct, kstep ks), lane L:
//   n = ct*16 + (L&15); element pair v holds K = ks*32 + (v>>2)*16 + (L>>4)*8 + 2(v&3), +1.
// Zero-padded past Kdim/Nout.
// ---------------------------------------------------------------------------
__global__ void k_pack_w(const float* __restrict__ W, _Float16* __restrict__ Bp,
                         int Kdim, int Nout, int ldw, int nColTiles, int ksteps) {
    int t = blockIdx.x * blockDim.x + threadIdx.x;
    int total = nColTiles * ksteps * 32 * 8;
    if (t >= total) return;
    int v    = t & 7;
    int lane = (t >> 3) & 31;
    int rest = t >> 8;
    int ks   = rest % ksteps;
    int ct   = rest / ksteps;
    int nloc = lane & 15;
    int hi   = lane >> 4;
    int k    = (ks << 5) + ((v >> 2) << 4) + (hi << 3) + ((v & 3) << 1);
    int n    = (ct << 4) + nloc;
    float w0 = (k     < Kdim && n < Nout) ? W[(long long)k       * ldw + n] : 0.0f;
    float w1 = (k + 1 < Kdim && n < Nout) ? W[(long long)(k + 1) * ldw + n] : 0.0f;
    long long base = (((long long)(ct * ksteps + ks) * 32 + lane) << 4) + (v << 1);
    Bp[base]     = (_Float16)w0;
    Bp[base + 1] = (_Float16)w1;
}

// ---------------------------------------------------------------------------
// WMMA GEMM: H[N x ldh] = (RELU ? relu(X) : X)[N x KDIM] @ Wpacked
// One wave per 16-row block; NCT 16-wide col tiles register-blocked so the
// A fragment is loaded once and reused across all col tiles.
// A fragment (16-bit A 16x32 layout): per lane two contiguous 8-float runs:
//   elems 0..7  <- K = ks*32 + hi*8 + [0..8)
//   elems 8..15 <- K = ks*32 + 16 + hi*8 + [0..8)
// Guards on runs are wave-uniform and const-fold after full unrolling.
// C layout: VGPR j -> M = 8*hi + j, N = lane&15.
// ---------------------------------------------------------------------------
template <int KDIM, int KSTEPS, int NCT, int RELU>
__global__ void k_gemm_wmma(const float* __restrict__ X, const _Float16* __restrict__ Bp,
                            float* __restrict__ H, int nRowTiles, int ldx, int ldh) {
    const int lane    = threadIdx.x & 31;
    const int rowTile = (int)(((long long)blockIdx.x * blockDim.x + threadIdx.x) >> 5);
    if (rowTile >= nRowTiles) return;
    const int r  = lane & 15;
    const int hi = lane >> 4;
    const float* __restrict__ xrow = X + (long long)((rowTile << 4) + r) * ldx;

    v8f acc[NCT];
#pragma unroll
    for (int ct = 0; ct < NCT; ++ct) acc[ct] = (v8f){};

#pragma unroll
    for (int ks = 0; ks < KSTEPS; ++ks) {
        const int kb = (ks << 5) + (hi << 3);
        float4 ra0{}, ra1{}, rb0{}, rb1{};
        if (ks * 32 + 15 < KDIM) {                 // run A fully in-range (const-folds)
            ra0 = *(const float4*)(xrow + kb);
            ra1 = *(const float4*)(xrow + kb + 4);
        }
        if (ks * 32 + 31 < KDIM) {                 // run B fully in-range (const-folds)
            rb0 = *(const float4*)(xrow + kb + 16);
            rb1 = *(const float4*)(xrow + kb + 20);
        }
        auto cv = [](float f) -> _Float16 {
            if (RELU) f = fmaxf(f, 0.0f);
            return (_Float16)f;
        };
        v16h a;
        a[0]  = cv(ra0.x); a[1]  = cv(ra0.y); a[2]  = cv(ra0.z); a[3]  = cv(ra0.w);
        a[4]  = cv(ra1.x); a[5]  = cv(ra1.y); a[6]  = cv(ra1.z); a[7]  = cv(ra1.w);
        a[8]  = cv(rb0.x); a[9]  = cv(rb0.y); a[10] = cv(rb0.z); a[11] = cv(rb0.w);
        a[12] = cv(rb1.x); a[13] = cv(rb1.y); a[14] = cv(rb1.z); a[15] = cv(rb1.w);

#pragma unroll
        for (int ct = 0; ct < NCT; ++ct) {
            const v16h b =
                *(const v16h*)(Bp + (((long long)(ct * KSTEPS + ks) * 32 + lane) << 4));
            acc[ct] = __builtin_amdgcn_wmma_f32_16x16x32_f16(false, a, false, b, (short)0,
                                                             acc[ct], false, false);
        }
    }

#pragma unroll
    for (int ct = 0; ct < NCT; ++ct) {
        float* __restrict__ hout =
            H + (long long)((rowTile << 4) + (hi << 3)) * ldh + (ct << 4) + r;
#pragma unroll
        for (int j = 0; j < 8; ++j)
            hout[(long long)j * ldh] = acc[ct][j];
    }
}

// ---------------------------------------------------------------------------
// Accumulator init: out[n,f] = H[n,f] * dis[n]^2 + bias[f]  (self-loop + bias)
// ---------------------------------------------------------------------------
template <int SHIFT, int DIMV>
__global__ void k_self_bias(const float* __restrict__ H, int ldh,
                            const float* __restrict__ bias, const float* __restrict__ dis,
                            float* __restrict__ out, int ldo, int n) {
    long long t = (long long)blockIdx.x * blockDim.x + threadIdx.x;
    int node = (int)(t >> SHIFT);
    int f    = (int)(t & ((1 << SHIFT) - 1));
    if (node >= n || f >= DIMV) return;
    float d = dis[node];
    out[(long long)node * ldo + f] = H[(long long)node * ldh + f] * d * d + bias[f];
}

// ---------------------------------------------------------------------------
// Edge scatter: out[dst] += H[src] * dis[src]*dis[dst]
// (1<<LSHIFT) lanes per edge, each lane: one float4 gather + 4 f32 atomics.
// ---------------------------------------------------------------------------
template <int LSHIFT, int DIMV>
__global__ void k_edge_scatter(const float* __restrict__ H, int ldh,
                               float* __restrict__ out, int ldo,
                               const int* __restrict__ src, const int* __restrict__ dst,
                               const float* __restrict__ dis, int nEdges) {
    long long t = (long long)blockIdx.x * blockDim.x + threadIdx.x;
    int e = (int)(t >> LSHIFT);
    if (e >= nEdges) return;
    int chunk = (int)(t & ((1 << LSHIFT) - 1));
    int s = src[e];
    int d = dst[e];
    float w = dis[s] * dis[d];
    int f0 = chunk << 2;
    const float4 v = *(const float4*)(H + (long long)s * ldh + f0);
    float* __restrict__ o = out + (long long)d * ldo;
    if (f0 + 0 < DIMV) atomicAdd(o + f0 + 0, v.x * w);
    if (f0 + 1 < DIMV) atomicAdd(o + f0 + 1, v.y * w);
    if (f0 + 2 < DIMV) atomicAdd(o + f0 + 2, v.z * w);
    if (f0 + 3 < DIMV) atomicAdd(o + f0 + 3, v.w * w);
}

// ---------------------------------------------------------------------------
// Launch
// ---------------------------------------------------------------------------
extern "C" void kernel_launch(void* const* d_in, const int* in_sizes, int n_in,
                              void* d_out, int out_size, void* d_ws, size_t ws_size,
                              hipStream_t stream) {
    const int IN_DIM = 16, HID = 128, OUT_DIM = 15;

    const float* x   = (const float*)d_in[0];
    const int*   ei  = (const int*)d_in[1];
    const float* W1  = (const float*)d_in[2];
    const float* b1  = (const float*)d_in[3];
    const float* W2  = (const float*)d_in[4];
    const float* b2  = (const float*)d_in[5];
    const float* W3  = (const float*)d_in[6];
    const float* b3  = (const float*)d_in[7];
    float* out = (float*)d_out;

    const int N = in_sizes[0] / IN_DIM;   // 100000
    const int E = in_sizes[1] / 2;        // 1600000
    const int* src = ei;
    const int* dst = ei + E;

    // ---- workspace partition (256B aligned) ----
    auto al = [](size_t v) { return (v + 255) & ~(size_t)255; };
    char* ws = (char*)d_ws;
    size_t o = 0;
    float* dis = (float*)(ws + o);       o += al((size_t)N * 4);
    float* Hb  = (float*)(ws + o);       o += al((size_t)N * HID * 4);
    float* Ab  = (float*)(ws + o);       o += al((size_t)N * HID * 4);
    _Float16* Bp1 = (_Float16*)(ws + o); o += al((size_t)8 * 1 * 512 * 2);
    _Float16* Bp2 = (_Float16*)(ws + o); o += al((size_t)8 * 4 * 512 * 2);
    _Float16* Bp3 = (_Float16*)(ws + o); o += al((size_t)1 * 4 * 512 * 2);
    (void)ws_size; (void)n_in; (void)out_size;

    auto cdiv = [](long long a, long long b) { return (int)((a + b - 1) / b); };
    const int BLK = 256;

    // ---- degrees / symmetric normalization ----
    k_init_deg<<<cdiv(N, BLK), BLK, 0, stream>>>(dis, N);
    k_edge_deg<<<cdiv(E, BLK), BLK, 0, stream>>>(dst, dis, E);
    k_rsqrt  <<<cdiv(N, BLK), BLK, 0, stream>>>(dis, N);

    // ---- pack weights into WMMA B fragments (zero-padded K/N) ----
    k_pack_w<<<cdiv(8 * 1 * 256, BLK), BLK, 0, stream>>>(W1, Bp1, IN_DIM, HID, HID, 8, 1);
    k_pack_w<<<cdiv(8 * 4 * 256, BLK), BLK, 0, stream>>>(W2, Bp2, HID, HID, HID, 8, 4);
    k_pack_w<<<cdiv(1 * 4 * 256, BLK), BLK, 0, stream>>>(W3, Bp3, HID, OUT_DIM, OUT_DIM, 1, 4);

    const int rowTiles = N / 16;                    // 6250 (exact)
    const int gemmBlocks = cdiv(rowTiles, 8);       // 8 waves (row tiles) per block

    // ---- layer 1: H = x @ W1 ; A = H*dis^2 + b1 ; scatter ----
    k_gemm_wmma<16, 1, 8, 0><<<gemmBlocks, BLK, 0, stream>>>(
        x, Bp1, Hb, rowTiles, IN_DIM, HID);
    k_self_bias<7, 128><<<cdiv((long long)N << 7, BLK), BLK, 0, stream>>>(
        Hb, HID, b1, dis, Ab, HID, N);
    k_edge_scatter<5, 128><<<cdiv((long long)E << 5, BLK), BLK, 0, stream>>>(
        Hb, HID, Ab, HID, src, dst, dis, E);

    // ---- layer 2: H = relu(A) @ W2 ; A = H*dis^2 + b2 ; scatter ----
    k_gemm_wmma<128, 4, 8, 1><<<gemmBlocks, BLK, 0, stream>>>(
        Ab, Bp2, Hb, rowTiles, HID, HID);
    k_self_bias<7, 128><<<cdiv((long long)N << 7, BLK), BLK, 0, stream>>>(
        Hb, HID, b2, dis, Ab, HID, N);
    k_edge_scatter<5, 128><<<cdiv((long long)E << 5, BLK), BLK, 0, stream>>>(
        Hb, HID, Ab, HID, src, dst, dis, E);

    // ---- layer 3: H = relu(A) @ W3 (padded to 16 cols) ; out = agg + b3 ----
    k_gemm_wmma<128, 4, 1, 1><<<gemmBlocks, BLK, 0, stream>>>(
        Ab, Bp3, Hb, rowTiles, HID, 16);
    k_self_bias<4, 15><<<cdiv((long long)N << 4, BLK), BLK, 0, stream>>>(
        Hb, 16, b3, dis, out, OUT_DIM, N);
    k_edge_scatter<2, 15><<<cdiv((long long)E << 2, BLK), BLK, 0, stream>>>(
        Hb, 16, out, OUT_DIM, src, dst, dis, E);
}
